// multihead_attention_43765716746559
// MI455X (gfx1250) — matrix operand order repaired and checked by reading, not compiled, not run on hardware
//
#include <hip/hip_runtime.h>

// ---------------------------------------------------------------------------
// Multi-head attention forward for MI455X (gfx1250, wave32, WMMA).
//   q = query@Wq^T, k = key@Wk^T, v = value@Wv^T      (bf16 operands, f32 acc)
//   flash-attention (online softmax) over 64-key blocks per (batch, head)
//   out = att@Wo^T + bo                                (fp32 output)
// All WMMA fragment reads are 16B-aligned b128 accesses.
// ---------------------------------------------------------------------------

#define S_LEN   2048
#define D_MODEL 1024
#define HEADS   16
#define D_K     64
#define NBATCH  2
#define M_TOT   (NBATCH * S_LEN)   // 4096 rows

typedef __attribute__((ext_vector_type(16))) __bf16        v16bf;
typedef __attribute__((ext_vector_type(8)))  float         v8f;
typedef __attribute__((ext_vector_type(4)))  float         v4f;
typedef __attribute__((ext_vector_type(4)))  unsigned int  v4u;

union BF16Frag {
    v16bf          v;
    unsigned short u[16];
    v4u            x[2];   // two 16B chunks
};

__device__ __forceinline__ unsigned short f2bf(float f) {
    unsigned u = __builtin_bit_cast(unsigned, f);
    unsigned r = u + 0x7FFFu + ((u >> 16) & 1u);   // round-to-nearest-even
    return (unsigned short)(r >> 16);
}

__device__ __forceinline__ unsigned int pack_bf2(float lo, float hi) {
    return (unsigned int)f2bf(lo) | ((unsigned int)f2bf(hi) << 16);
}

__device__ __forceinline__ v8f wmma_bf16(const BF16Frag& a, const BF16Frag& b, v8f c) {
    return __builtin_amdgcn_wmma_f32_16x16x32_bf16(
        /*neg_a=*/false, a.v, /*neg_b=*/false, b.v,
        /*c_mod=*/(short)0, c, /*reuse_a=*/false, /*reuse_b=*/false);
}

// Load a fragment as two 16B LDS/global reads (addresses are 16B aligned).
__device__ __forceinline__ void frag_ld(BF16Frag& f, const unsigned short* p0,
                                        const unsigned short* p1) {
    f.x[0] = *(const v4u*)p0;
    f.x[1] = *(const v4u*)p1;
}

// ---------------------------------------------------------------------------
// QKV projection: out[m,n] = sum_k X[m,k] * W[n,k]   (Linear, W is [out,in])
// MODE 0/1 : store bf16 at [b, h, s, d]   (Q and K)
// MODE 2   : store bf16 at [b, h, d, s]   (V transposed -> PV B-frags are rows)
// Block: 128 threads (4 waves). Block tile 128(M) x 64(N); each wave owns a
// 32x64 sub-tile (2 A-frags share each B-frag -> 8 WMMA per 32-K step).
// ---------------------------------------------------------------------------
template <int MODE>
__global__ __launch_bounds__(128)
void proj_qkv_kernel(const float* __restrict__ X, const float* __restrict__ W,
                     unsigned short* __restrict__ out) {
    // row strides (40, in ushorts) = 80B = 16B*5 : frag reads stay 16B aligned
    __shared__ __attribute__((aligned(16))) unsigned short As[128][40];
    __shared__ __attribute__((aligned(16))) unsigned short Bs[64][40];

    const int t = threadIdx.x;
    const int w = t >> 5, lane = t & 31;
    const int halfsel = lane >> 4, ln = lane & 15;
    const int mBase = blockIdx.y * 128, nBase = blockIdx.x * 64;

    v8f acc[2][4] = {{{}, {}, {}, {}}, {{}, {}, {}, {}}};

    for (int kk = 0; kk < D_MODEL; kk += 32) {
        // stage X tile 128x32 as bf16 (float4 loads, packed bf16 stores)
        for (int i = t; i < 128 * 8; i += 128) {
            const int r = i >> 3, c = i & 7;
            v4f xv = *(const v4f*)&X[(size_t)(mBase + r) * D_MODEL + kk + c * 4];
            unsigned int* p = (unsigned int*)&As[r][c * 4];
            p[0] = pack_bf2(xv[0], xv[1]);
            p[1] = pack_bf2(xv[2], xv[3]);
        }
        // stage W tile 64x32 as bf16
        for (int i = t; i < 64 * 8; i += 128) {
            const int r = i >> 3, c = i & 7;
            v4f wv = *(const v4f*)&W[(size_t)(nBase + r) * D_MODEL + kk + c * 4];
            unsigned int* p = (unsigned int*)&Bs[r][c * 4];
            p[0] = pack_bf2(wv[0], wv[1]);
            p[1] = pack_bf2(wv[2], wv[3]);
        }
        __syncthreads();

        BF16Frag a0, a1;
        const int arow0 = w * 32 + ln, arow1 = arow0 + 16;
        const int kof = halfsel * 8;
        frag_ld(a0, &As[arow0][kof], &As[arow0][16 + kof]);
        frag_ld(a1, &As[arow1][kof], &As[arow1][16 + kof]);
#pragma unroll
        for (int nt = 0; nt < 4; ++nt) {
            BF16Frag b;
            const int brow = nt * 16 + ln, kb = halfsel * 8;
            frag_ld(b, &Bs[brow][kb], &Bs[brow][kb + 16]);
            acc[0][nt] = wmma_bf16(a0, b, acc[0][nt]);
            acc[1][nt] = wmma_bf16(a1, b, acc[1][nt]);
        }
        __syncthreads();
    }

#pragma unroll
    for (int mi = 0; mi < 2; ++mi) {
#pragma unroll
        for (int nt = 0; nt < 4; ++nt) {
#pragma unroll
            for (int r = 0; r < 8; ++r) {
                const int m = mBase + w * 32 + mi * 16 + r + halfsel * 8;
                const int n = nBase + nt * 16 + ln;
                const int b = m >> 11, s = m & (S_LEN - 1);
                const int h = n >> 6, d = n & (D_K - 1);
                const unsigned short val = f2bf(acc[mi][nt][r]);
                if (MODE == 2)
                    out[((size_t)(b * HEADS + h) * D_K + d) * S_LEN + s] = val;
                else
                    out[((size_t)(b * HEADS + h) * S_LEN + s) * D_K + d] = val;
            }
        }
    }
}

// ---------------------------------------------------------------------------
// Flash attention. Grid: (S/64 q-blocks, NBATCH*HEADS). Block: 128 = 4 waves.
// Each wave owns 16 query rows; streams 64-key blocks with online softmax.
// ---------------------------------------------------------------------------
__global__ __launch_bounds__(128)
void attn_kernel(const unsigned short* __restrict__ Q,
                 const unsigned short* __restrict__ K,
                 const unsigned short* __restrict__ Vt,
                 const int* __restrict__ mask,
                 unsigned short* __restrict__ att) {
    // stride 72 ushorts = 144B = 16B*9 -> frag reads 16B aligned, conflict-free
    __shared__ __attribute__((aligned(16))) unsigned short Plds[4][16][72];

    const int t = threadIdx.x;
    const int w = t >> 5, lane = t & 31;
    const int halfsel = lane >> 4, ln = lane & 15;
    const int bh = blockIdx.y;
    const int b = bh >> 4, h = bh & 15;
    const int qBase = blockIdx.x * 64 + w * 16;

    const unsigned short* Qhead = Q  + (size_t)(b * HEADS + h) * S_LEN * D_K;
    const unsigned short* Khead = K  + (size_t)(b * HEADS + h) * S_LEN * D_K;
    const unsigned short* Vhead = Vt + (size_t)(b * HEADS + h) * D_K * S_LEN;

    // Load this wave's Q rows as two A-fragments (K-dim = d_k = 64 = 2x32).
    BF16Frag aq[2];
    {
        const unsigned short* qrow = Qhead + (size_t)(qBase + ln) * D_K;
        const int kof = halfsel * 8;
#pragma unroll
        for (int ks = 0; ks < 2; ++ks)
            frag_ld(aq[ks], qrow + ks * 32 + kof, qrow + ks * 32 + 16 + kof);
    }

    v8f o[4] = {{}, {}, {}, {}};
    float mrow[8], lrow[8];
#pragma unroll
    for (int r = 0; r < 8; ++r) { mrow[r] = -1e30f; lrow[r] = 0.0f; }
    const float scale = 0.125f;            // 1/sqrt(64)
    const float negmask = -1e10f * scale;  // reference masks then scales

    for (int kb = 0; kb < S_LEN; kb += 64) {
        // ---- S = Q K^T for a 16x64 tile -------------------------------
        v8f sacc[4] = {{}, {}, {}, {}};
#pragma unroll
        for (int ks = 0; ks < 2; ++ks) {
#pragma unroll
            for (int nt = 0; nt < 4; ++nt) {
                BF16Frag bk;
                const unsigned short* krow =
                    Khead + (size_t)(kb + nt * 16 + ln) * D_K + ks * 32 + halfsel * 8;
                frag_ld(bk, krow, krow + 16);
                sacc[nt] = wmma_bf16(aq[ks], bk, sacc[nt]);
            }
        }

        // ---- mask + scale --------------------------------------------
#pragma unroll
        for (int nt = 0; nt < 4; ++nt) {
            const int kcol = kb + nt * 16 + ln;
#pragma unroll
            for (int r = 0; r < 8; ++r) {
                const int qrow_i = qBase + r + halfsel * 8;
                const int mval = mask[((size_t)b * S_LEN + qrow_i) * S_LEN + kcol];
                sacc[nt][r] = (mval == 0) ? negmask : sacc[nt][r] * scale;
            }
        }

        // ---- online softmax (row = register r within each half-wave) --
#pragma unroll
        for (int r = 0; r < 8; ++r) {
            float tmax = fmaxf(fmaxf(sacc[0][r], sacc[1][r]),
                               fmaxf(sacc[2][r], sacc[3][r]));
#pragma unroll
            for (int off = 1; off < 16; off <<= 1)
                tmax = fmaxf(tmax, __shfl_xor(tmax, off, 32));
            const float nm = fmaxf(mrow[r], tmax);
            const float alpha = __expf(mrow[r] - nm);
            mrow[r] = nm;
            float tsum = 0.0f;
#pragma unroll
            for (int nt = 0; nt < 4; ++nt) {
                const float p = __expf(sacc[nt][r] - nm);
                sacc[nt][r] = p;
                tsum += p;
            }
#pragma unroll
            for (int off = 1; off < 16; off <<= 1)
                tsum += __shfl_xor(tsum, off, 32);
            lrow[r] = lrow[r] * alpha + tsum;
#pragma unroll
            for (int dt = 0; dt < 4; ++dt) o[dt][r] *= alpha;
        }

        // ---- re-layout P (C-layout -> A-layout) via per-wave LDS ------
#pragma unroll
        for (int nt = 0; nt < 4; ++nt)
#pragma unroll
            for (int r = 0; r < 8; ++r)
                Plds[w][r + halfsel * 8][nt * 16 + ln] = f2bf(sacc[nt][r]);
        __syncthreads();

        // ---- O += P V ------------------------------------------------
#pragma unroll
        for (int ks = 0; ks < 2; ++ks) {
            BF16Frag ap;
            const int kof = halfsel * 8;
            frag_ld(ap, &Plds[w][ln][ks * 32 + kof], &Plds[w][ln][ks * 32 + 16 + kof]);
#pragma unroll
            for (int dt = 0; dt < 4; ++dt) {
                BF16Frag bv;
                const unsigned short* vrow =
                    Vhead + (size_t)(dt * 16 + ln) * S_LEN + kb + ks * 32 + halfsel * 8;
                frag_ld(bv, vrow, vrow + 16);
                o[dt] = wmma_bf16(ap, bv, o[dt]);
            }
        }
        __syncthreads();
    }

    // ---- finalize: divide by row sums, write concat-head bf16 --------
#pragma unroll
    for (int dt = 0; dt < 4; ++dt)
#pragma unroll
        for (int r = 0; r < 8; ++r) {
            const int q = qBase + r + halfsel * 8;
            const float val = o[dt][r] / lrow[r];
            att[((size_t)(b * S_LEN + q)) * D_MODEL + h * D_K + dt * 16 + ln] =
                f2bf(val);
        }
}

// ---------------------------------------------------------------------------
// Output projection: out = att(bf16) @ Wo^T + bo   (fp32 out)
// Same 128x64 block tile as proj_qkv, A already bf16 (b128 copy into LDS).
// ---------------------------------------------------------------------------
__global__ __launch_bounds__(128)
void out_proj_kernel(const unsigned short* __restrict__ Abf,
                     const float* __restrict__ W,
                     const float* __restrict__ bias,
                     float* __restrict__ out) {
    __shared__ __attribute__((aligned(16))) unsigned short As[128][40];
    __shared__ __attribute__((aligned(16))) unsigned short Bs[64][40];

    const int t = threadIdx.x;
    const int w = t >> 5, lane = t & 31;
    const int halfsel = lane >> 4, ln = lane & 15;
    const int mBase = blockIdx.y * 128, nBase = blockIdx.x * 64;

    v8f acc[2][4] = {{{}, {}, {}, {}}, {{}, {}, {}, {}}};

    for (int kk = 0; kk < D_MODEL; kk += 32) {
        for (int i = t; i < 128 * 4; i += 128) {       // 8-ushort (16B) chunks
            const int r = i >> 2, c = i & 3;
            *(v4u*)&As[r][c * 8] =
                *(const v4u*)&Abf[(size_t)(mBase + r) * D_MODEL + kk + c * 8];
        }
        for (int i = t; i < 64 * 8; i += 128) {
            const int r = i >> 3, c = i & 7;
            v4f wv = *(const v4f*)&W[(size_t)(nBase + r) * D_MODEL + kk + c * 4];
            unsigned int* p = (unsigned int*)&Bs[r][c * 4];
            p[0] = pack_bf2(wv[0], wv[1]);
            p[1] = pack_bf2(wv[2], wv[3]);
        }
        __syncthreads();

        BF16Frag a0, a1;
        const int arow0 = w * 32 + ln, arow1 = arow0 + 16;
        const int kof = halfsel * 8;
        frag_ld(a0, &As[arow0][kof], &As[arow0][16 + kof]);
        frag_ld(a1, &As[arow1][kof], &As[arow1][16 + kof]);
#pragma unroll
        for (int nt = 0; nt < 4; ++nt) {
            BF16Frag b;
            const int brow = nt * 16 + ln, kb = halfsel * 8;
            frag_ld(b, &Bs[brow][kb], &Bs[brow][kb + 16]);
            acc[0][nt] = wmma_bf16(a0, b, acc[0][nt]);
            acc[1][nt] = wmma_bf16(a1, b, acc[1][nt]);
        }
        __syncthreads();
    }

#pragma unroll
    for (int mi = 0; mi < 2; ++mi)
#pragma unroll
        for (int nt = 0; nt < 4; ++nt)
#pragma unroll
            for (int r = 0; r < 8; ++r) {
                const int m = mBase + w * 32 + mi * 16 + r + halfsel * 8;
                const int n = nBase + nt * 16 + ln;
                out[(size_t)m * D_MODEL + n] = acc[mi][nt][r] + bias[n];
            }
}

// ---------------------------------------------------------------------------
extern "C" void kernel_launch(void* const* d_in, const int* in_sizes, int n_in,
                              void* d_out, int out_size, void* d_ws, size_t ws_size,
                              hipStream_t stream) {
    (void)in_sizes; (void)n_in; (void)out_size; (void)ws_size;

    const float* value = (const float*)d_in[0];
    const float* key   = (const float*)d_in[1];
    const float* query = (const float*)d_in[2];
    const int*   mask  = (const int*)d_in[3];
    const float* Wq    = (const float*)d_in[4];
    const float* Wk    = (const float*)d_in[5];
    const float* Wv    = (const float*)d_in[6];
    const float* Wo    = (const float*)d_in[7];
    const float* bo    = (const float*)d_in[8];
    float* out = (float*)d_out;

    unsigned short* ws = (unsigned short*)d_ws;
    const size_t PER = (size_t)NBATCH * HEADS * S_LEN * D_K;  // 4,194,304 elems
    unsigned short* Qb = ws;              // [b,h,s,d]  bf16
    unsigned short* Kb = ws + PER;        // [b,h,s,d]  bf16
    unsigned short* Vb = ws + 2 * PER;    // [b,h,d,s]  bf16 (transposed)
    unsigned short* Ab = ws + 3 * PER;    // [b*s, d_model] bf16

    dim3 blk(128);
    dim3 gproj(D_MODEL / 64, M_TOT / 128);
    proj_qkv_kernel<0><<<gproj, blk, 0, stream>>>(query, Wq, Qb);
    proj_qkv_kernel<1><<<gproj, blk, 0, stream>>>(key,   Wk, Kb);
    proj_qkv_kernel<2><<<gproj, blk, 0, stream>>>(value, Wv, Vb);

    dim3 gattn(S_LEN / 64, NBATCH * HEADS);
    attn_kernel<<<gattn, blk, 0, stream>>>(Qb, Kb, Vb, mask, Ab);

    out_proj_kernel<<<gproj, blk, 0, stream>>>(Ab, Wo, bo, out);
}
